// DCN_Conv_70866960384166
// MI455X (gfx1250) — matrix operand order repaired and checked by reading, not compiled, not run on hardware
//
#include <hip/hip_runtime.h>
#include <hip/hip_bf16.h>
#include <math.h>

typedef __attribute__((ext_vector_type(4)))  _Float16 v4h;
typedef __attribute__((ext_vector_type(8)))  _Float16 v8h;
typedef __attribute__((ext_vector_type(16))) _Float16 v16h;
typedef __attribute__((ext_vector_type(8)))  float    v8f;

#define Dd   64
#define Wd   64
#define Hd   64
#define SP   (Dd*Wd*Hd)       // 262144
#define KK   9
#define CIN  8
#define COUT 16
#define OFFC 18               // only z/y offset channels are consumed by the reference
#define KDIM (CIN*KK)         // 72  (K order: k = kk*8 + cin)
#define KPAD 96               // pad K to 3 x 32 for wmma_f32_16x16x32_f16

// ---- workspace layout (float offsets) ----
#define BN_SUM   0
#define BN_SSQ   32
#define BN_SCALE 64
#define BN_SHIFT 96
#define G_SUM    128
#define G_SSQ    160
#define GN_SCALE 192
#define GN_SHIFT 224
#define OFFS_RAW 256
#define PREGN    (256 + OFFC*SP)

// ------------------------------------------------------------------
// K1: zero the stat accumulators (must run every launch: graph replays)
// ------------------------------------------------------------------
__global__ void zero_stats_kernel(float* __restrict__ ws) {
    ws[threadIdx.x] = 0.0f;   // 256 floats cover all stat slots
}

// ------------------------------------------------------------------
// K2: offset conv (3x3x3, pad 1) for the 18 used channels + BN stats
// grid = (SP/256, OFFC), block = 256
// ------------------------------------------------------------------
__global__ __launch_bounds__(256) void offset_conv_kernel(
    const float* __restrict__ f, const float* __restrict__ off_w,
    const float* __restrict__ off_b, float* __restrict__ ws) {
    __shared__ float rs[256], rss[256];
    const int c   = blockIdx.y;
    const int tid = threadIdx.x;
    const int s   = blockIdx.x * 256 + tid;
    const int z = s >> 12, y = (s >> 6) & 63, x = s & 63;

    float acc = off_b[c];
    for (int ci = 0; ci < CIN; ++ci) {
        const float* fp = f + ci * SP;
        const float* wp = off_w + (c * CIN + ci) * 27;
        #pragma unroll
        for (int dz = -1; dz <= 1; ++dz) {
            const int zz = z + dz; if ((unsigned)zz > 63u) continue;
            #pragma unroll
            for (int dy = -1; dy <= 1; ++dy) {
                const int yy = y + dy; if ((unsigned)yy > 63u) continue;
                #pragma unroll
                for (int dx = -1; dx <= 1; ++dx) {
                    const int xx = x + dx; if ((unsigned)xx > 63u) continue;
                    acc += fp[zz * 4096 + yy * 64 + xx] *
                           wp[(dz + 1) * 9 + (dy + 1) * 3 + (dx + 1)];
                }
            }
        }
    }
    ws[OFFS_RAW + c * SP + s] = acc;

    // block reduce sum / sumsq for BatchNorm stats
    rs[tid] = acc; rss[tid] = acc * acc;
    __syncthreads();
    for (int st = 128; st > 0; st >>= 1) {
        if (tid < st) { rs[tid] += rs[tid + st]; rss[tid] += rss[tid + st]; }
        __syncthreads();
    }
    if (tid == 0) {
        atomicAdd(&ws[BN_SUM + c], rs[0]);
        atomicAdd(&ws[BN_SSQ + c], rss[0]);
    }
}

// ------------------------------------------------------------------
// K3: BN mean/var -> affine scale/shift
// ------------------------------------------------------------------
__global__ void finalize_bn_kernel(const float* __restrict__ bn_g,
                                   const float* __restrict__ bn_b,
                                   float* __restrict__ ws) {
    const int c = threadIdx.x;
    if (c < OFFC) {
        const float n    = (float)SP;
        const float mean = ws[BN_SUM + c] / n;
        const float var  = ws[BN_SSQ + c] / n - mean * mean;
        const float inv  = rsqrtf(var + 1e-5f);
        ws[BN_SCALE + c] = bn_g[c] * inv;
        ws[BN_SHIFT + c] = bn_b[c] - bn_g[c] * inv * mean;
    }
}

// ------------------------------------------------------------------
// K4: main kernel. Per wave: 16 x-positions at fixed (z,y).
//   phase A: bn+tanh offsets, cumsum-from-center -> coords (LDS) + prefetch
//   phase B: trilinear gather -> f16 A tile in LDS (K = kk*8+cin, padded 96)
//   phase C: 3x v_wmma_f32_16x16x32_f16 (b128 fragment loads), bias,
//            b128 stores of preGN, GroupNorm partial stats
// grid = 2048 blocks x 256 threads (8 waves)
// ------------------------------------------------------------------
__global__ __launch_bounds__(256) void dcn_wmma_kernel(
    const float* __restrict__ f, const float* __restrict__ dcn_w,
    const float* __restrict__ dcn_b, float* __restrict__ ws) {
    alignas(16) __shared__ _Float16 Atile[8][16][KPAD];   // 24576 B
    alignas(16) __shared__ _Float16 Bf[2][3][16][16];     //  3072 B, fragment-ready
    __shared__ float cz[8][16][KK];                       //  4608 B
    __shared__ float cy[8][16][KK];                       //  4608 B
    __shared__ float gacc[8];                             // [group*2 + {sum,ssq}]

    const int tid  = threadIdx.x;
    const int w    = tid >> 5;     // wave in block
    const int lane = tid & 31;
    const int m    = lane & 15;    // spatial row / out-channel col
    const int half = lane >> 4;

    const int wg = blockIdx.x * 8 + w;        // 16384 waves total
    const int x0 = (wg & 3) * 16;
    const int y  = (wg >> 2) & 63;
    const int z  = wg >> 8;

    // cooperative fill of fragment-ready B: Bf[half][c][n][j] = w^T[k][n],
    // k = 32*c + half*16 + j, source index w[n][cin][kk] with k = kk*8+cin.
    for (int i = tid; i < 2 * 3 * 16 * 16; i += 256) {
        const int hf = i / 768;
        const int r  = i - hf * 768;
        const int c  = r >> 8;
        const int n  = (r >> 4) & 15;
        const int j  = r & 15;
        const int k  = 32 * c + hf * 16 + j;
        Bf[hf][c][n][j] = (k < KDIM)
            ? (_Float16)dcn_w[n * KDIM + (k & 7) * KK + (k >> 3)]
            : (_Float16)0.0f;
    }
    if (tid < 8) gacc[tid] = 0.0f;
    __syncthreads();

    // ---- phase A ----
    if (half == 0) {
        const int sp = z * 4096 + y * 64 + (x0 + m);
        float t[OFFC];
        #pragma unroll
        for (int c = 0; c < OFFC; ++c) {
            const float o = ws[OFFS_RAW + c * SP + sp];
            t[c] = tanhf(o * ws[BN_SCALE + c] + ws[BN_SHIFT + c]);
        }
        float az[KK], ay[KK];
        az[4] = t[4]; ay[4] = t[13];
        #pragma unroll
        for (int j = 5; j < KK; ++j) { az[j] = az[j-1] + t[j]; ay[j] = ay[j-1] + t[9+j]; }
        #pragma unroll
        for (int j = 3; j >= 0; --j) { az[j] = az[j+1] + t[j]; ay[j] = ay[j+1] + t[9+j]; }
        #pragma unroll
        for (int j = 0; j < KK; ++j) {
            const float zcf = (float)z + az[j];
            const float ycf = (float)y + ay[j];
            cz[w][m][j] = zcf;
            cy[w][m][j] = ycf;
            // head-start for the gather phase: global_prefetch_b8
            const int pz = min(max((int)zcf, 0), 63);
            const int py = min(max((int)ycf, 0), 63);
            __builtin_prefetch(f + pz * 4096 + py * 64 + (x0 + m), 0, 0);
            __builtin_prefetch(f + 4 * SP + pz * 4096 + py * 64 + (x0 + m), 0, 0);
        }
    }
    __syncthreads();

    // ---- phase B ----
    {
        const int   cin0  = half * 4;
        const float xbase = (float)(x0 + m);
        #pragma unroll
        for (int kk = 0; kk < KK; ++kk) {
            const float zc = cz[w][m][kk];
            const float yc = cy[w][m][kk];
            const float xc = xbase + (float)(kk - 4);
            const float fz = floorf(zc), fy = floorf(yc), fx = floorf(xc);
            const int z0 = min(max((int)fz, 0), 63), z1 = min(max((int)fz + 1, 0), 63);
            const int y0 = min(max((int)fy, 0), 63), y1 = min(max((int)fy + 1, 0), 63);
            const int xa = min(max((int)fx, 0), 63), xb = min(max((int)fx + 1, 0), 63);
            const float wz0 = (float)z1 - zc, wz1 = zc - (float)z0;
            const float wy0 = (float)y1 - yc, wy1 = yc - (float)y0;
            const float wx0 = (float)xb - xc, wx1 = xc - (float)xa;
            const float w000 = wz0*wy0*wx0, w001 = wz0*wy0*wx1;
            const float w010 = wz0*wy1*wx0, w011 = wz0*wy1*wx1;
            const float w100 = wz1*wy0*wx0, w101 = wz1*wy0*wx1;
            const float w110 = wz1*wy1*wx0, w111 = wz1*wy1*wx1;
            const int i000 = z0*4096 + y0*64 + xa, i001 = z0*4096 + y0*64 + xb;
            const int i010 = z0*4096 + y1*64 + xa, i011 = z0*4096 + y1*64 + xb;
            const int i100 = z1*4096 + y0*64 + xa, i101 = z1*4096 + y0*64 + xb;
            const int i110 = z1*4096 + y1*64 + xa, i111 = z1*4096 + y1*64 + xb;
            float vs[4];
            #pragma unroll
            for (int c = 0; c < 4; ++c) {
                const float* fp = f + (cin0 + c) * SP;
                vs[c] = fp[i000]*w000 + fp[i001]*w001 + fp[i010]*w010 + fp[i011]*w011
                      + fp[i100]*w100 + fp[i101]*w101 + fp[i110]*w110 + fp[i111]*w111;
            }
            v4h pk;
            pk[0] = (_Float16)vs[0]; pk[1] = (_Float16)vs[1];
            pk[2] = (_Float16)vs[2]; pk[3] = (_Float16)vs[3];
            *(v4h*)&Atile[w][m][kk * 8 + cin0] = pk;        // 8B-aligned ds_store_b64
        }
        // zero K padding 72..95 (12 halves per lane half, 8B-aligned stores)
        const v4h z4 = {};
        #pragma unroll
        for (int j = 0; j < 3; ++j)
            *(v4h*)&Atile[w][m][KDIM + half * 12 + j * 4] = z4;
    }
    __syncthreads();

    // ---- phase C: D = A(16x96) x B(96x16), f16 in / f32 acc ----
    v8f acc = {};
    #pragma unroll
    for (int c = 0; c < 3; ++c) {
        // A layout: lane m holds K 0..15; lane m+16 holds K 16..31
        const v8h a0 = *(const v8h*)&Atile[w][m][32 * c + half * 16];
        const v8h a1 = *(const v8h*)&Atile[w][m][32 * c + half * 16 + 8];
        // B layout: lanes 0-15 hold K 0..15; lanes 16-31 hold K 16..31
        const v8h b0 = *(const v8h*)&Bf[half][c][m][0];
        const v8h b1 = *(const v8h*)&Bf[half][c][m][8];
        const v16h a = __builtin_shufflevector(a0, a1, 0,1,2,3,4,5,6,7,8,9,10,11,12,13,14,15);
        const v16h b = __builtin_shufflevector(b0, b1, 0,1,2,3,4,5,6,7,8,9,10,11,12,13,14,15);
        acc = __builtin_amdgcn_wmma_f32_16x16x32_f16(false, a, false, b,
                                                     (short)0, acc, false, false);
    }

    // bias + store 8 consecutive x positions (two b128 stores) + GN partials
    {
        const int   co   = m;                     // N index
        const float bias = dcn_b[co];
        float* preg = ws + PREGN + co * SP + z * 4096 + y * 64 + x0 + half * 8;
        float vv[8], s = 0.0f, ss = 0.0f;
        #pragma unroll
        for (int r = 0; r < 8; ++r) { vv[r] = acc[r] + bias; s += vv[r]; ss += vv[r]*vv[r]; }
        *(float4*)(preg)     = make_float4(vv[0], vv[1], vv[2], vv[3]);
        *(float4*)(preg + 4) = make_float4(vv[4], vv[5], vv[6], vv[7]);
        const int g = co >> 2;
        atomicAdd(&gacc[g * 2 + 0], s);
        atomicAdd(&gacc[g * 2 + 1], ss);
    }
    __syncthreads();
    if (tid < 8) {
        const int g = tid >> 1, which = tid & 1;
        atomicAdd(&ws[(which ? G_SSQ : G_SUM) + g], gacc[g * 2 + which]);
    }
}

// ------------------------------------------------------------------
// K5: GroupNorm mean/var -> per-channel affine
// ------------------------------------------------------------------
__global__ void finalize_gn_kernel(const float* __restrict__ gn_g,
                                   const float* __restrict__ gn_b,
                                   float* __restrict__ ws) {
    const int c = threadIdx.x;
    if (c < COUT) {
        const int   g    = c >> 2;
        const float n    = (float)(4 * SP);
        const float mean = ws[G_SUM + g] / n;
        const float var  = ws[G_SSQ + g] / n - mean * mean;
        const float inv  = rsqrtf(var + 1e-5f);
        ws[GN_SCALE + c] = gn_g[c] * inv;
        ws[GN_SHIFT + c] = gn_b[c] - gn_g[c] * inv * mean;
    }
}

// ------------------------------------------------------------------
// K6: normalize + ReLU, vectorized float4
// ------------------------------------------------------------------
__global__ __launch_bounds__(256) void apply_gn_relu_kernel(
    const float* __restrict__ ws, float* __restrict__ out) {
    const int i4 = blockIdx.x * 256 + threadIdx.x;     // COUT*SP/4 elements
    const int e  = i4 * 4;
    const int c  = e >> 18;                            // / SP
    const float sc = ws[GN_SCALE + c], sh = ws[GN_SHIFT + c];
    const float4 v = *(const float4*)(ws + PREGN + e);
    float4 o;
    o.x = fmaxf(v.x * sc + sh, 0.0f);
    o.y = fmaxf(v.y * sc + sh, 0.0f);
    o.z = fmaxf(v.z * sc + sh, 0.0f);
    o.w = fmaxf(v.w * sc + sh, 0.0f);
    *(float4*)(out + e) = o;
}

// ------------------------------------------------------------------
extern "C" void kernel_launch(void* const* d_in, const int* in_sizes, int n_in,
                              void* d_out, int out_size, void* d_ws, size_t ws_size,
                              hipStream_t stream) {
    (void)in_sizes; (void)n_in; (void)out_size; (void)ws_size;
    const float* f     = (const float*)d_in[0];
    const float* off_w = (const float*)d_in[1];
    const float* off_b = (const float*)d_in[2];
    const float* bn_g  = (const float*)d_in[3];
    const float* bn_b  = (const float*)d_in[4];
    const float* dcn_w = (const float*)d_in[5];
    const float* dcn_b = (const float*)d_in[6];
    const float* gn_g  = (const float*)d_in[7];
    const float* gn_b  = (const float*)d_in[8];
    float* ws  = (float*)d_ws;
    float* out = (float*)d_out;

    zero_stats_kernel<<<1, 256, 0, stream>>>(ws);
    offset_conv_kernel<<<dim3(SP / 256, OFFC), 256, 0, stream>>>(f, off_w, off_b, ws);
    finalize_bn_kernel<<<1, 32, 0, stream>>>(bn_g, bn_b, ws);
    dcn_wmma_kernel<<<16384 / 8, 256, 0, stream>>>(f, dcn_w, dcn_b, ws);
    finalize_gn_kernel<<<1, 32, 0, stream>>>(gn_g, gn_b, ws);
    apply_gn_relu_kernel<<<COUT * SP / 4 / 256, 256, 0, stream>>>(ws, out);
}
